// BitLinearV2_10995116278314
// MI455X (gfx1250) — 8-bit matrix operand order repaired and checked by reading, compiled, not run on hardware
//
#include <hip/hip_runtime.h>
#include <cstdint>
#include <cstddef>


typedef __attribute__((ext_vector_type(8))) int   v8i;
typedef __attribute__((ext_vector_type(4))) int   v4i;
typedef __attribute__((ext_vector_type(2))) int   v2i;
typedef __attribute__((ext_vector_type(4))) float v4f;

#define IN_F  4096
#define OUT_F 11008
#define MTOT  8192

#if defined(__has_builtin)
#if __has_builtin(__builtin_amdgcn_global_load_async_to_lds_b128)
#define USE_ASYNC_LDS 1
#endif
#endif
#ifndef USE_ASYNC_LDS
#define USE_ASYNC_LDS 0
#endif

__device__ __forceinline__ void wait_async0() {
#if defined(__has_builtin) && __has_builtin(__builtin_amdgcn_s_wait_asynccnt)
  __builtin_amdgcn_s_wait_asynccnt(0);
#else
  asm volatile("s_wait_asynccnt 0x0" ::: "memory");
#endif
}

__device__ __forceinline__ int clamp127(int q) {
  return q > 127 ? 127 : (q < -127 ? -127 : q);
}

__global__ __launch_bounds__(256) void quant_act(const float* __restrict__ x,
                                                 signed char* __restrict__ q,
                                                 float* __restrict__ ascale)
{
  const int m = blockIdx.x;
  const int t = threadIdx.x;
  const v4f* row4 = (const v4f*)(x + (size_t)m * IN_F);

  v4f v[4];
  float amax = 0.f;
#pragma unroll
  for (int j = 0; j < 4; ++j) {
    v[j] = row4[t + j * 256];
    amax = fmaxf(amax, fmaxf(fmaxf(fabsf(v[j].x), fabsf(v[j].y)),
                             fmaxf(fabsf(v[j].z), fabsf(v[j].w))));
  }

  __shared__ float red[256];
  red[t] = amax;
  __syncthreads();
  for (int off = 128; off > 0; off >>= 1) {
    if (t < off) red[t] = fmaxf(red[t], red[t + off]);
    __syncthreads();
  }
  amax = red[0];

  const float inv = (amax > 0.f) ? 127.f / amax : 0.f;
  if (t == 0) ascale[m] = (amax > 0.f) ? amax / 127.f : 0.f;

  int* q4 = (int*)(q + (size_t)m * IN_F);
#pragma unroll
  for (int j = 0; j < 4; ++j) {
    const int q0 = clamp127(__float2int_rn(v[j].x * inv));
    const int q1 = clamp127(__float2int_rn(v[j].y * inv));
    const int q2 = clamp127(__float2int_rn(v[j].z * inv));
    const int q3 = clamp127(__float2int_rn(v[j].w * inv));
    q4[t + j * 256] = (q0 & 255) | ((q1 & 255) << 8) | ((q2 & 255) << 16) | ((q3 & 255) << 24);
  }
}

__global__ __launch_bounds__(256) void tern_w(const float* __restrict__ w,
                                              const float* __restrict__ th,
                                              int* __restrict__ wq)
{
  const int idx = blockIdx.x * 256 + threadIdx.x;
  const v4f v = ((const v4f*)w)[idx];
  const float t = th[idx >> 10];
  const int q0 = (fabsf(v.x) >= t) ? (v.x > 0.f ? 1 : -1) : 0;
  const int q1 = (fabsf(v.y) >= t) ? (v.y > 0.f ? 1 : -1) : 0;
  const int q2 = (fabsf(v.z) >= t) ? (v.z > 0.f ? 1 : -1) : 0;
  const int q3 = (fabsf(v.w) >= t) ? (v.w > 0.f ? 1 : -1) : 0;
  wq[idx] = (q0 & 255) | ((q1 & 255) << 8) | ((q2 & 255) << 16) | ((q3 & 255) << 24);
}

#define BM 128
#define BN 128
#define KC 64
#define LDK 80
#define NSTAGE (IN_F / KC)

__global__ __launch_bounds__(256) void bitlinear_gemm(
    const signed char* __restrict__ Aq,
    const signed char* __restrict__ Wt,
    const float* __restrict__ ascale,
    const float* __restrict__ wscale,
    const float* __restrict__ bias,
    float* __restrict__ out)
{
  __shared__ signed char sA[2][BM * LDK];
  __shared__ signed char sB[2][BN * LDK];

  const int tid   = threadIdx.x;
  const int lane  = tid & 31;
  const int wave  = tid >> 5;
  const int waveM = wave >> 2;
  const int waveN = wave & 3;
  const int m0 = blockIdx.y * BM;
  const int n0 = blockIdx.x * BN;

  const int lrow = tid >> 2;
  const int lcol = (tid & 3) * 16;

  v8i acc[4][2] = {};

  const int rA  = lane & 15;
  const int hiA = (lane >> 4) << 4;
  const int hiB = (lane >> 4) << 4;

  auto compute = [&](int buf) {
    v8i bf[2];
#pragma unroll
    for (int ni = 0; ni < 2; ++ni) {
      const signed char* p = sB[buf] + (waveN * 32 + ni * 16 + rA) * LDK + hiB;
      const v4i b0 = *(const v4i*)(p);
      const v4i b1 = *(const v4i*)(p + 32);
      bf[ni] = __builtin_shufflevector(b0, b1, 0, 1, 2, 3, 4, 5, 6, 7);
    }
#pragma unroll
    for (int mi = 0; mi < 4; ++mi) {
      const signed char* p = sA[buf] + (waveM * 64 + mi * 16 + rA) * LDK + hiA;
      const v2i a0 = *(const v2i*)(p);
      const v2i a1 = *(const v2i*)(p + 8);
      const v2i a2 = *(const v2i*)(p + 32);
      const v2i a3 = *(const v2i*)(p + 40);
      v8i af = { a0.x, a0.y, a1.x, a1.y, a2.x, a2.y, a3.x, a3.y };
      acc[mi][0] = __builtin_amdgcn_wmma_i32_16x16x64_iu8(
          true, af, true, bf[0], acc[mi][0], false, false);
      acc[mi][1] = __builtin_amdgcn_wmma_i32_16x16x64_iu8(
          true, af, true, bf[1], acc[mi][1], false, false);
    }
  };

#if USE_ASYNC_LDS
  auto stage_async = [&](int kk, int buf) {
#pragma unroll
    for (int j = 0; j < 2; ++j) {
      const int r = lrow + j * 64;
      __builtin_amdgcn_global_load_async_to_lds_b128(
          (v4i*)(Aq + (size_t)(m0 + r) * IN_F + kk + lcol),
          (v4i*)(sA[buf] + r * LDK + lcol), 0, 0);
      __builtin_amdgcn_global_load_async_to_lds_b128(
          (v4i*)(Wt + (size_t)(n0 + r) * IN_F + kk + lcol),
          (v4i*)(sB[buf] + r * LDK + lcol), 0, 0);
    }
  };

  stage_async(0, 0);
  wait_async0();
  __syncthreads();
  for (int s = 0; s < NSTAGE; s += 2) {
    stage_async((s + 1) * KC, 1);
    compute(0);
    wait_async0();
    __syncthreads();

    if (s + 2 < NSTAGE) stage_async((s + 2) * KC, 0);
    compute(1);
    if (s + 2 < NSTAGE) wait_async0();
    __syncthreads();
  }
#else
  v4i pa[2], pb[2];
  auto prefetch = [&](int kk) {
#pragma unroll
    for (int j = 0; j < 2; ++j) {
      const int r = lrow + j * 64;
      pa[j] = *(const v4i*)(Aq + (size_t)(m0 + r) * IN_F + kk + lcol);
      pb[j] = *(const v4i*)(Wt + (size_t)(n0 + r) * IN_F + kk + lcol);
    }
  };
  auto commit = [&](int buf) {
#pragma unroll
    for (int j = 0; j < 2; ++j) {
      const int r = lrow + j * 64;
      *(v4i*)(sA[buf] + r * LDK + lcol) = pa[j];
      *(v4i*)(sB[buf] + r * LDK + lcol) = pb[j];
    }
  };

  prefetch(0);
  commit(0);
  for (int s = 0; s < NSTAGE; s += 2) {
    __syncthreads();
    prefetch((s + 1) * KC);
    compute(0);
    commit(1);
    __syncthreads();
    if (s + 2 < NSTAGE) prefetch((s + 2) * KC);
    compute(1);
    if (s + 2 < NSTAGE) commit(0);
  }
#endif

  const int oBase = n0 + waveN * 32 + (lane & 15);
  const float sc0 = wscale[oBase],      bv0 = bias[oBase];
  const float sc1 = wscale[oBase + 16], bv1 = bias[oBase + 16];
  const int mBase = m0 + waveM * 64 + ((lane >> 4) << 3);
#pragma unroll
  for (int mi = 0; mi < 4; ++mi) {
#pragma unroll
    for (int r = 0; r < 8; ++r) {
      const int m = mBase + mi * 16 + r;
      const float as = ascale[m];
      float* po = out + (size_t)m * OUT_F + oBase;
      po[0]  = as * sc0 * (float)acc[mi][0][r] + bv0;
      po[16] = as * sc1 * (float)acc[mi][1][r] + bv1;
    }
  }
}

__global__ void fallback_gemm(const float* __restrict__ x, const float* __restrict__ w,
                              const float* __restrict__ sc, const float* __restrict__ th,
                              const float* __restrict__ bs, float* __restrict__ out,
                              size_t total)
{
  const size_t i = (size_t)blockIdx.x * blockDim.x + threadIdx.x;
  if (i >= total) return;
  const int o = (int)(i % OUT_F);
  const size_t m = i / OUT_F;
  const float* xr = x + m * IN_F;
  const float* wr = w + (size_t)o * IN_F;
  const float t = th[o];
  float acc = 0.f;
  for (int k = 0; k < IN_F; ++k) {
    const float wv = wr[k];
    const float qv = (fabsf(wv) >= t) ? (wv > 0.f ? 1.f : -1.f) : 0.f;
    acc += xr[k] * qv;
  }
  out[i] = acc * sc[o] + bs[o];
}

extern "C" void kernel_launch(void* const* d_in, const int* in_sizes, int n_in,
                              void* d_out, int out_size, void* d_ws, size_t ws_size,
                              hipStream_t stream)
{
  const float* x  = (const float*)d_in[0];
  const float* w  = (const float*)d_in[1];
  const float* sc = (const float*)d_in[2];
  const float* th = (const float*)d_in[3];
  const float* bs = (const float*)d_in[4];
  float* out = (float*)d_out;

  const size_t bytesA = (size_t)MTOT * IN_F;
  const size_t bytesW = (size_t)OUT_F * IN_F;
  const size_t need   = bytesA + bytesW + (size_t)MTOT * 4;

  if (ws_size < need) {
    const size_t total = (size_t)MTOT * OUT_F;
    const int blocks = (int)((total + 255) / 256);
    fallback_gemm<<<blocks, 256, 0, stream>>>(x, w, sc, th, bs, out, total);
    return;
  }

  signed char* Aq = (signed char*)d_ws;
  signed char* Wt = Aq + bytesA;
  float* asc = (float*)(Aq + bytesA + bytesW);

  quant_act<<<MTOT, 256, 0, stream>>>(x, Aq, asc);
  tern_w<<<(int)((bytesW / 4) / 256), 256, 0, stream>>>(w, th, (int*)Wt);
  bitlinear_gemm<<<dim3(OUT_F / BN, MTOT / BM), 256, 0, stream>>>(Aq, Wt, asc, sc, bs, out);
}
